// STGNN_85822036509295
// MI455X (gfx1250) — matrix operand order repaired and checked by reading, not compiled, not run on hardware
//
#include <hip/hip_runtime.h>
#include <hip/hip_bf16.h>
#include <math.h>

typedef __attribute__((ext_vector_type(16))) __bf16 bf16x16;
typedef __attribute__((ext_vector_type(8)))  float  f32x8;
typedef __attribute__((ext_vector_type(4)))  int    v4i;

#define Hh   4
#define Cc   64
#define HID  256
#define NTn  50000
#define NCn  10000
#define ETT  400000
#define ECT  200000
#define FTf  128
#define FCf  64

// ===========================================================================
// Packed fragment layouts (wave32 WMMA, per CDNA5 ISA 7.12.2):
//  A-pack:  [mtile][kstep][lane][e], lane = (row%16) + 16*half,
//           K(e,half) = ((e&8)<<1) | (e&7) | (half<<3)       (16-bit A 16x32)
//  B-pack:  [ntile][kstep][lane][e], col = lane%16,
//           K(e,half) = ((e&8)<<1) | (e&7) | (half<<3)       (16-bit B 32x16)
// Each lane's fragment is 16 contiguous bf16 (32B) -> 2x global_load_b128.
// ===========================================================================

// pack activations A[M][K] (f32) -> A-fragment order (bf16)
__global__ void pack_act_k(const float* __restrict__ A, __bf16* __restrict__ out,
                           int M, int KS)   // KS = K/32
{
    int gid = blockIdx.x * blockDim.x + threadIdx.x;
    if (gid >= M * (KS << 5)) return;        // M*K elements
    // decompose gid = ((mt*KS + kt)*32 + lane)*16 + e
    int e    = gid & 15;
    int lane = (gid >> 4) & 31;
    int rest = gid >> 9;
    int kt = rest % KS, mt = rest / KS;
    int half = lane >> 4, l16 = lane & 15;
    int kl = ((e & 8) << 1) | (e & 7) | (half << 3);
    int K = KS * 32;
    out[gid] = (__bf16)A[(size_t)(mt * 16 + l16) * K + kt * 32 + kl];
}

// pack weights W[K][N] (f32) -> B-fragment order (bf16)
__global__ void pack_weight_k(const float* __restrict__ W, __bf16* __restrict__ out,
                              int KS, int N)  // KS = K/32
{
    int gid = blockIdx.x * blockDim.x + threadIdx.x;
    if (gid >= (KS << 5) * N) return;        // K*N elements
    int e    = gid & 15;
    int lane = (gid >> 4) & 31;
    int rest = gid >> 9;
    int kt = rest % KS, nt = rest / KS;
    int half = lane >> 4, l16 = lane & 15;
    out[gid] = (__bf16)W[(size_t)(kt * 32 + (((e & 8) << 1) | (e & 7) | (half << 3))) * N + nt * 16 + l16];
}

// packed A index for an element at (row, col) of a matrix whose columns are
// the next GEMM's K dimension (KSn = cols/32)
__device__ inline size_t pk_index(int row, int col, int KSn)
{
    int mt = row >> 4, rl = row & 15;
    int kt = col >> 5, kl = col & 31;
    int half2 = (kl >> 3) & 1;
    int e = ((kl & 16) >> 1) | (kl & 7);
    return (((size_t)mt * KSn + kt) * 32 + rl + 16 * half2) * 16 + e;
}

// ===========================================================================
// GEMM: C[M,N=256] = act(A @ W + bias); A,B pre-packed bf16; optional f32 C
// and packed-bf16 shadow Cpk (fragment order for the next GEMM, KSnext=N/32).
// Block 128 thr = 4 waves; block tile 16(M) x 256(N); each wave: 4 n-tiles.
// K loop unrolled x2 with explicit ping-pong fragment sets: one set computes
// while the other set's loads are in flight (no rotation copies).
// Requires KS even (true here: 4, 2, 8).
// ===========================================================================
__global__ __launch_bounds__(128)
void gemm_pk_wmma(const __bf16* __restrict__ Ap, const __bf16* __restrict__ Bp,
                  const float* __restrict__ bias, float* __restrict__ Cp,
                  __bf16* __restrict__ Cpk,
                  int M, int KS, int N, int KSnext, int doRelu, int hasBias)
{
    const int mt   = blockIdx.x;
    const int t    = threadIdx.x;
    const int lane = t & 31;
    const int wave = t >> 5;
    const int half = lane >> 4, l16 = lane & 15;
    const int nt0  = wave * 4;               // N==256: 16 n-tiles over 4 waves

    const __bf16* a = Ap + ((size_t)mt  * KS * 32 + lane) * 16;
    const __bf16* b = Bp + ((size_t)nt0 * KS * 32 + lane) * 16;
    const size_t bstride = (size_t)KS * 512;

    f32x8 acc[4] = {{}, {}, {}, {}};

    bf16x16 afA, bfA[4], afB, bfB[4];
    afA = *reinterpret_cast<const bf16x16*>(a);
    #pragma unroll
    for (int s = 0; s < 4; ++s)
        bfA[s] = *reinterpret_cast<const bf16x16*>(b + (size_t)s * bstride);

    for (int kt = 0; kt < KS; kt += 2) {
        // load set B (kt+1) while set A (kt) computes
        a += 512; b += 512;
        afB = *reinterpret_cast<const bf16x16*>(a);
        #pragma unroll
        for (int s = 0; s < 4; ++s)
            bfB[s] = *reinterpret_cast<const bf16x16*>(b + (size_t)s * bstride);

        #pragma unroll
        for (int s = 0; s < 4; ++s)
            acc[s] = __builtin_amdgcn_wmma_f32_16x16x32_bf16(
                false, afA, false, bfA[s], (short)0, acc[s], false, false);

        // load set A (kt+2) while set B computes (last pair: benign reload)
        int adv = (kt + 2 < KS) ? 512 : 0;
        a += adv; b += adv;
        if (kt + 2 < KS) __builtin_prefetch(a + 512, 0, 0);
        afA = *reinterpret_cast<const bf16x16*>(a);
        #pragma unroll
        for (int s = 0; s < 4; ++s)
            bfA[s] = *reinterpret_cast<const bf16x16*>(b + (size_t)s * bstride);

        #pragma unroll
        for (int s = 0; s < 4; ++s)
            acc[s] = __builtin_amdgcn_wmma_f32_16x16x32_bf16(
                false, afB, false, bfB[s], (short)0, acc[s], false, false);
    }

    #pragma unroll
    for (int r = 0; r < 8; ++r) {
        int row = mt * 16 + half * 8 + r;
        #pragma unroll
        for (int s = 0; s < 4; ++s) {
            int col = (nt0 + s) * 16 + l16;
            float v = acc[s][r];
            if (hasBias) v += bias[col];
            if (doRelu)  v = fmaxf(v, 0.0f);
            if (Cp)  Cp[(size_t)row * N + col] = v;
            if (Cpk) Cpk[pk_index(row, col, KSnext)] = (__bf16)v;
        }
    }
}

// ===========================================================================
// s[n,h] = sum_c p[n, h*64+c] * att[h*64+c]; att staged in LDS via the CDNA5
// async-to-LDS path when available (global_load_async_to_lds_b128 +
// s_wait_asynccnt), else plain loads.
// ===========================================================================
__global__ __launch_bounds__(256)
void att_reduce_k(const float* __restrict__ p, const float* __restrict__ att,
                  float* __restrict__ out, int Nnodes)
{
    __shared__ alignas(16) float atts[HID];
#if __has_builtin(__builtin_amdgcn_global_load_async_to_lds_b128)
    if (threadIdx.x < 64) {
        __builtin_amdgcn_global_load_async_to_lds_b128(
            (__attribute__((address_space(1))) v4i*)(att + threadIdx.x * 4),
            (__attribute__((address_space(3))) v4i*)(&atts[threadIdx.x * 4]),
            0, 0);
    }
  #if __has_builtin(__builtin_amdgcn_s_wait_asynccnt)
    __builtin_amdgcn_s_wait_asynccnt(0);
  #else
    asm volatile("s_wait_asynccnt 0x0" ::: "memory");
  #endif
#else
    if (threadIdx.x < HID) atts[threadIdx.x] = att[threadIdx.x];
#endif
    __syncthreads();

    int i = blockIdx.x * blockDim.x + threadIdx.x;  // over Nnodes*H
    if (i >= Nnodes * Hh) return;
    int n = i >> 2, h = i & 3;
    const float4* pp = reinterpret_cast<const float4*>(p + (size_t)n * HID + h * Cc);
    const float4* aa = reinterpret_cast<const float4*>(&atts[h * Cc]);
    float s = 0.0f;
    #pragma unroll
    for (int q = 0; q < 16; ++q) {
        float4 x = pp[q], w = aa[q];
        s += x.x * w.x + x.y * w.y + x.z * w.z + x.w * w.w;
    }
    out[i] = s;
}

__global__ void fill_f32_k(float* p, float v, int n) {
    int i = blockIdx.x * blockDim.x + threadIdx.x;
    if (i < n) p[i] = v;
}

// float atomic-max via int/uint trick (valid with -inf init)
__device__ inline void atomicMaxFloat(float* addr, float val) {
    if (val >= 0.0f) atomicMax((int*)addr, __float_as_int(val));
    else             atomicMin((unsigned int*)addr, (unsigned int)__float_as_int(val));
}

// ===========================================================================
// GAT edge pass 1: a = leaky_relu(s[src]+d[dst], 0.2); m[dst] = max(m[dst], a)
// Edges e >= Emain are self-loops (node = e - Emain).
// ===========================================================================
__global__ void gat_edge_max_k(const int* __restrict__ srcI, const int* __restrict__ dstI,
                               int Emain, int Etot,
                               const float* __restrict__ s, const float* __restrict__ d,
                               float* __restrict__ a_buf, float* __restrict__ m)
{
    int gid = blockIdx.x * blockDim.x + threadIdx.x;
    if (gid >= Etot * Hh) return;
    int e = gid >> 2, h = gid & 3;
    int si, di;
    if (e < Emain) { si = srcI[e]; di = dstI[e]; } else { si = di = e - Emain; }
    float a = s[si * Hh + h] + d[di * Hh + h];
    a = (a > 0.0f) ? a : 0.2f * a;
    a_buf[gid] = a;
    atomicMaxFloat(&m[di * Hh + h], a);
}

// ===========================================================================
// GAT edge pass 2: ex = exp(a - mfix[dst]); msg[dst,:] += ex*xs[src,:]; den += ex
// One 256-thread block per edge.
// ===========================================================================
__global__ __launch_bounds__(256)
void gat_edge_scatter_k(const int* __restrict__ srcI, const int* __restrict__ dstI,
                        int Emain, int Etot,
                        const float* __restrict__ a_buf, const float* __restrict__ m,
                        const float* __restrict__ xs,
                        float* __restrict__ msg, float* __restrict__ den)
{
    int e = blockIdx.x;
    int t = threadIdx.x;
    int h = t >> 6;
    int si, di;
    if (e < Emain) { si = srcI[e]; di = dstI[e]; } else { si = di = e - Emain; }
    float mm = m[di * Hh + h];
    if (!__builtin_isfinite(mm)) mm = 0.0f;
    float ex = __expf(a_buf[e * Hh + h] - mm);
    atomicAdd(&msg[(size_t)di * HID + t], ex * xs[(size_t)si * HID + t]);
    if ((t & 63) == 0) atomicAdd(&den[di * Hh + h], ex);
}

// ===========================================================================
// combine: x = relu(0.25*(o_s2d + o_d2s) + 0.5*o_ct + ht), written directly
// in packed bf16 A-fragment order (KSnext = HID/32 = 8) for the final GEMM.
// ===========================================================================
__global__ void combine_k(const float* __restrict__ msg1, const float* __restrict__ den1, const float* __restrict__ b1,
                          const float* __restrict__ msg2, const float* __restrict__ den2, const float* __restrict__ b2,
                          const float* __restrict__ msg3, const float* __restrict__ den3, const float* __restrict__ b3,
                          const float* __restrict__ ht, __bf16* __restrict__ xpk)
{
    int gid = blockIdx.x * blockDim.x + threadIdx.x;
    if (gid >= NTn * HID) return;
    int n = gid >> 8, t = gid & 255, h = t >> 6;
    float o1 = msg1[gid] / fmaxf(den1[n * Hh + h], 1e-16f) + b1[t];
    float o2 = msg2[gid] / fmaxf(den2[n * Hh + h], 1e-16f) + b2[t];
    float o3 = msg3[gid] / fmaxf(den3[n * Hh + h], 1e-16f) + b3[t];
    float v = 0.25f * (o1 + o2) + 0.5f * o3 + ht[gid];
    xpk[pk_index(n, t, HID / 32)] = (__bf16)fmaxf(v, 0.0f);
}

// ===========================================================================
// host launch
// ===========================================================================
static inline void gemm(const __bf16* Ap, const __bf16* Bp, const float* bias,
                        float* C, __bf16* Cpk, int M, int K, int N,
                        bool relu, hipStream_t stream)
{
    dim3 grid(M / 16, 1);                    // N == 256 handled by one block col
    gemm_pk_wmma<<<grid, 128, 0, stream>>>(Ap, Bp, bias, C, Cpk,
                                           M, K / 32, N, N / 32,
                                           relu ? 1 : 0, bias ? 1 : 0);
}

extern "C" void kernel_launch(void* const* d_in, const int* in_sizes, int n_in,
                              void* d_out, int out_size, void* d_ws, size_t ws_size,
                              hipStream_t stream)
{
    (void)in_sizes; (void)n_in; (void)out_size; (void)ws_size;

    const float* x_target  = (const float*)d_in[0];
    const float* x_context = (const float*)d_in[1];
    const int*   ei_tt     = (const int*)d_in[2];     // [2, ETT]
    const int*   ei_ct_src = (const int*)d_in[3];
    const int*   ei_ct_dst = (const int*)d_in[4];
    const float* Wt = (const float*)d_in[5];  const float* bt = (const float*)d_in[6];
    const float* Wc = (const float*)d_in[7];  const float* bc = (const float*)d_in[8];
    const float* W_s2d = (const float*)d_in[9];
    const float* as_s2d = (const float*)d_in[10]; const float* ad_s2d = (const float*)d_in[11];
    const float* b_s2d = (const float*)d_in[12];
    const float* W_d2s = (const float*)d_in[13];
    const float* as_d2s = (const float*)d_in[14]; const float* ad_d2s = (const float*)d_in[15];
    const float* b_d2s = (const float*)d_in[16];
    const float* W_ct_src = (const float*)d_in[17]; const float* W_ct_dst = (const float*)d_in[18];
    const float* as_ct = (const float*)d_in[19]; const float* ad_ct = (const float*)d_in[20];
    const float* b_ct = (const float*)d_in[21];
    const float* W_out = (const float*)d_in[22]; const float* b_out = (const float*)d_in[23];

    char* ws = (char*)d_ws;
    size_t off = 0;
    auto alloc = [&](size_t bytes) -> char* {
        char* p = ws + off;
        off += (bytes + 255) & ~(size_t)255;
        return p;
    };

    // packed bf16 weights
    __bf16* WtP   = (__bf16*)alloc((size_t)FTf * HID * 2);
    __bf16* WcP   = (__bf16*)alloc((size_t)FCf * HID * 2);
    __bf16* Ws2dP = (__bf16*)alloc((size_t)HID * HID * 2);
    __bf16* Wd2sP = (__bf16*)alloc((size_t)HID * HID * 2);
    __bf16* WctsP = (__bf16*)alloc((size_t)HID * HID * 2);
    __bf16* WctdP = (__bf16*)alloc((size_t)HID * HID * 2);
    __bf16* WoutP = (__bf16*)alloc((size_t)HID * HID * 2);

    // packed bf16 activations
    __bf16* xtP = (__bf16*)alloc((size_t)NTn * FTf * 2);
    __bf16* xcP = (__bf16*)alloc((size_t)NCn * FCf * 2);
    __bf16* htP = (__bf16*)alloc((size_t)NTn * HID * 2);
    __bf16* hcP = (__bf16*)alloc((size_t)NCn * HID * 2);

    // f32 activations
    float* ht = (float*)alloc((size_t)NTn * HID * 4);
    float* p1 = (float*)alloc((size_t)NTn * HID * 4);
    float* p2 = (float*)alloc((size_t)NTn * HID * 4);
    float* pd = (float*)alloc((size_t)NTn * HID * 4);
    float* ps = (float*)alloc((size_t)NCn * HID * 4);

    // per-node-head attention dots
    float* s1  = (float*)alloc((size_t)NTn * Hh * 4);
    float* d1  = (float*)alloc((size_t)NTn * Hh * 4);
    float* s2  = (float*)alloc((size_t)NTn * Hh * 4);
    float* d2  = (float*)alloc((size_t)NTn * Hh * 4);
    float* sct = (float*)alloc((size_t)NCn * Hh * 4);
    float* dct = (float*)alloc((size_t)NTn * Hh * 4);

    const int ETT_TOT = ETT + NTn;   // with self-loops
    float* a1 = (float*)alloc((size_t)ETT_TOT * Hh * 4);
    float* a2 = (float*)alloc((size_t)ETT_TOT * Hh * 4);
    float* a3 = (float*)alloc((size_t)ECT * Hh * 4);

    // m1..m3 contiguous for one -inf fill
    float* m1 = (float*)alloc((size_t)3 * NTn * Hh * 4);
    float* m2 = m1 + (size_t)NTn * Hh;
    float* m3 = m2 + (size_t)NTn * Hh;

    // msg1..3, den1..3 contiguous for one memset
    size_t msgden_bytes = (size_t)3 * NTn * HID * 4 + (size_t)3 * NTn * Hh * 4;
    float* msg1 = (float*)alloc(msgden_bytes);
    float* msg2 = msg1 + (size_t)NTn * HID;
    float* msg3 = msg2 + (size_t)NTn * HID;
    float* den1 = msg3 + (size_t)NTn * HID;
    float* den2 = den1 + (size_t)NTn * Hh;
    float* den3 = den2 + (size_t)NTn * Hh;

    __bf16* xpk = (__bf16*)p1;  // p1 dead by combine time; reuse its space

    // --- pack weights into B-fragment order ---
    auto packW = [&](const float* W, __bf16* dst, int K, int N) {
        int n = K * N;
        pack_weight_k<<<(n + 255) / 256, 256, 0, stream>>>(W, dst, K / 32, N);
    };
    packW(Wt, WtP, FTf, HID);
    packW(Wc, WcP, FCf, HID);
    packW(W_s2d, Ws2dP, HID, HID);
    packW(W_d2s, Wd2sP, HID, HID);
    packW(W_ct_src, WctsP, HID, HID);
    packW(W_ct_dst, WctdP, HID, HID);
    packW(W_out, WoutP, HID, HID);

    // --- pack input activations into A-fragment order ---
    {
        int n1 = NTn * FTf;
        pack_act_k<<<(n1 + 255) / 256, 256, 0, stream>>>(x_target, xtP, NTn, FTf / 32);
        int n2 = NCn * FCf;
        pack_act_k<<<(n2 + 255) / 256, 256, 0, stream>>>(x_context, xcP, NCn, FCf / 32);
    }

    // --- pretransform + relu (f32 out + packed bf16 shadow) ---
    gemm(xtP, WtP, bt, ht, htP, NTn, FTf, HID, true, stream);
    gemm(xcP, WcP, bc, nullptr, hcP, NCn, FCf, HID, true, stream);

    // --- projections (f32 only) ---
    gemm(htP, Ws2dP, nullptr, p1, nullptr, NTn, HID, HID, false, stream);
    gemm(htP, Wd2sP, nullptr, p2, nullptr, NTn, HID, HID, false, stream);
    gemm(htP, WctdP, nullptr, pd, nullptr, NTn, HID, HID, false, stream);
    gemm(hcP, WctsP, nullptr, ps, nullptr, NCn, HID, HID, false, stream);

    // --- attention dot reductions ---
    auto attred = [&](const float* p, const float* att, float* out, int Nn) {
        att_reduce_k<<<(Nn * Hh + 255) / 256, 256, 0, stream>>>(p, att, out, Nn);
    };
    attred(p1, as_s2d, s1, NTn);
    attred(p1, ad_s2d, d1, NTn);
    attred(p2, as_d2s, s2, NTn);
    attred(p2, ad_d2s, d2, NTn);
    attred(ps, as_ct, sct, NCn);
    attred(pd, ad_ct, dct, NTn);

    // --- init accumulators ---
    (void)hipMemsetAsync(msg1, 0, msgden_bytes, stream);
    fill_f32_k<<<(3 * NTn * Hh + 255) / 256, 256, 0, stream>>>(
        m1, -__builtin_inff(), 3 * NTn * Hh);

    const int* srcTT = ei_tt;
    const int* dstTT = ei_tt + ETT;

    // --- pass 1: s2d (forward edges) ---
    gat_edge_max_k<<<(ETT_TOT * Hh + 255) / 256, 256, 0, stream>>>(
        srcTT, dstTT, ETT, ETT_TOT, s1, d1, a1, m1);
    gat_edge_scatter_k<<<ETT_TOT, 256, 0, stream>>>(
        srcTT, dstTT, ETT, ETT_TOT, a1, m1, p1, msg1, den1);

    // --- pass 2: d2s (reversed edges) ---
    gat_edge_max_k<<<(ETT_TOT * Hh + 255) / 256, 256, 0, stream>>>(
        dstTT, srcTT, ETT, ETT_TOT, s2, d2, a2, m2);
    gat_edge_scatter_k<<<ETT_TOT, 256, 0, stream>>>(
        dstTT, srcTT, ETT, ETT_TOT, a2, m2, p2, msg2, den2);

    // --- pass 3: context -> target (no self-loops) ---
    gat_edge_max_k<<<(ECT * Hh + 255) / 256, 256, 0, stream>>>(
        ei_ct_src, ei_ct_dst, ECT, ECT, sct, dct, a3, m3);
    gat_edge_scatter_k<<<ECT, 256, 0, stream>>>(
        ei_ct_src, ei_ct_dst, ECT, ECT, a3, m3, ps, msg3, den3);

    // --- combine + skip + relu -> packed bf16 (overwrites p1 region) ---
    combine_k<<<(NTn * HID + 255) / 256, 256, 0, stream>>>(
        msg1, den1, b_s2d, msg2, den2, b_d2s, msg3, den3, b_ct, ht, xpk);

    // --- final linear into d_out ---
    gemm(xpk, WoutP, b_out, (float*)d_out, nullptr, NTn, HID, HID, false, stream);
}
